// HeteroGNNEncoder_83966610637468
// MI455X (gfx1250) — matrix operand order repaired and checked by reading, not compiled, not run on hardware
//
#include <hip/hip_runtime.h>
#include <hip/hip_bf16.h>
#include <math.h>

#define NVERT 50000
#define NEDGE 80000
#define NFACE 30000
#define EREL  160000
#define HDIM  256
#define NHEAD 4
#define CDIM  64
#define NLAYER 3
#define NGRAPH 128
#define LATD  64
#define TED   8

typedef __attribute__((ext_vector_type(16))) __bf16 v16bf;
typedef __attribute__((ext_vector_type(8)))  float  v8f;

// ---------------- utility ----------------
__global__ void k_fill(float* __restrict__ p, float v, long n) {
  long i = (long)blockIdx.x * blockDim.x + threadIdx.x;
  long st = (long)gridDim.x * blockDim.x;
  for (; i < n; i += st) p[i] = v;
}

__device__ __forceinline__ void atomicMaxF(float* addr, float val) {
  // monotonic int mapping for IEEE-754 floats
  if (val >= 0.0f) atomicMax((int*)addr, __float_as_int(val));
  else             atomicMin((unsigned int*)addr, (unsigned int)__float_as_int(val));
}

// ---------------- input projection + LayerNorm + ReLU ----------------
__global__ __launch_bounds__(256)
void k_proj_ln_relu(const float* __restrict__ X, int din,
                    const float* __restrict__ emb, const int* __restrict__ typ,
                    const float* __restrict__ W, const float* __restrict__ b,
                    const float* __restrict__ g, const float* __restrict__ beta,
                    float* __restrict__ out) {
  __shared__ float xr[32];
  __shared__ float red[HDIM];
  int n = blockIdx.x, c = threadIdx.x;
  int dtot = din + (emb ? TED : 0);
  if (c < din) xr[c] = X[(size_t)n * din + c];
  else if (emb && c < dtot) xr[c] = emb[typ[n] * TED + (c - din)];
  __syncthreads();
  float acc = b[c];
  for (int k = 0; k < dtot; ++k) acc += xr[k] * W[k * HDIM + c];
  red[c] = acc; __syncthreads();
  for (int s = 128; s > 0; s >>= 1) { if (c < s) red[c] += red[c + s]; __syncthreads(); }
  float mean = red[0] * (1.0f / HDIM);
  __syncthreads();
  float d = acc - mean;
  red[c] = d * d; __syncthreads();
  for (int s = 128; s > 0; s >>= 1) { if (c < s) red[c] += red[c + s]; __syncthreads(); }
  float var = red[0] * (1.0f / HDIM);
  float y = d * rsqrtf(var + 1e-5f) * g[c] + beta[c];
  out[(size_t)n * HDIM + c] = fmaxf(y, 0.0f);
}

// ---------------- residual + LayerNorm + ReLU update (in place) ----------------
__global__ __launch_bounds__(256)
void k_update_ln_relu(float* __restrict__ h, const float* __restrict__ msg,
                      const float* __restrict__ b1, const float* __restrict__ b2,
                      const float* __restrict__ g, const float* __restrict__ beta) {
  __shared__ float red[HDIM];
  int n = blockIdx.x, c = threadIdx.x;
  float acc = h[(size_t)n * HDIM + c] + msg[(size_t)n * HDIM + c] + b1[c];
  if (b2) acc += b2[c];
  red[c] = acc; __syncthreads();
  for (int s = 128; s > 0; s >>= 1) { if (c < s) red[c] += red[c + s]; __syncthreads(); }
  float mean = red[0] * (1.0f / HDIM);
  __syncthreads();
  float d = acc - mean;
  red[c] = d * d; __syncthreads();
  for (int s = 128; s > 0; s >>= 1) { if (c < s) red[c] += red[c + s]; __syncthreads(); }
  float var = red[0] * (1.0f / HDIM);
  float y = d * rsqrtf(var + 1e-5f) * g[c] + beta[c];
  h[(size_t)n * HDIM + c] = fmaxf(y, 0.0f);
}

// ---------------- pre-swizzle W[256,256] f32 -> bf16 B-fragment layout ----------------
// idx = ((kstep*16 + ntile)*32 + lane)*16 + j
// element j of lane's v16bf: K = kstep*32 + (lane>>4)*8 + (j&7) + (j>>3)*16 ; N = ntile*16 + (lane&15)
__global__ void k_swizzle_w(const float* __restrict__ W, __bf16* __restrict__ out) {
  int idx = blockIdx.x * blockDim.x + threadIdx.x;
  if (idx >= 65536) return;
  int j = idx & 15, lane = (idx >> 4) & 31, ntile = (idx >> 9) & 15, ks = idx >> 13;
  int k = ks * 32 + (lane >> 4) * 8 + (j & 7) + (j >> 3) * 16;
  int n = ntile * 16 + (lane & 15);
  out[idx] = (__bf16)W[k * HDIM + n];
}

// ---------------- collapse W[256,256] with attention vector a[4,64] -> M[256,4] ----------------
__global__ void k_collapse_aw(const float* __restrict__ W, const float* __restrict__ a,
                              float* __restrict__ out) {
  int t = blockIdx.x * blockDim.x + threadIdx.x;
  if (t >= HDIM * NHEAD) return;
  int k = t >> 2, h = t & 3;
  float acc = 0.0f;
  for (int c = 0; c < CDIM; ++c) acc += W[k * HDIM + h * CDIM + c] * a[h * CDIM + c];
  out[t] = acc; // [k*4 + h]
}

// ---------------- thin GEMM: out[n,4] = X[n,256] @ M[256,4] (+bias) ----------------
__global__ void k_thin_gemm4(const float* __restrict__ X, const float* __restrict__ M,
                             const float* __restrict__ bias, float* __restrict__ out, int n) {
  int t = blockIdx.x * blockDim.x + threadIdx.x;
  if (t >= n * NHEAD) return;
  int row = t >> 2, h = t & 3;
  const float* xr = X + (size_t)row * HDIM;
  float acc = bias ? bias[h] : 0.0f;
  for (int k = 0; k < HDIM; ++k) acc += xr[k] * M[k * NHEAD + h];
  out[t] = acc;
}

// ---------------- WMMA GEMM fragment loaders ----------------
__device__ __forceinline__ v16bf ldA(const float* __restrict__ arow, int ks, int khalf, bool rowok) {
  v16bf af;
  if (rowok) {
    const float* p0 = arow + ks * 32 + khalf * 8;  // K run 0 (8 contiguous f32)
    const float* p1 = p0 + 16;                     // K run 1
    #pragma unroll
    for (int j = 0; j < 8; ++j) af[j] = (__bf16)p0[j];
    #pragma unroll
    for (int j = 0; j < 8; ++j) af[8 + j] = (__bf16)p1[j];
  } else {
    #pragma unroll
    for (int j = 0; j < 16; ++j) af[j] = (__bf16)0.0f;
  }
  return af;
}

__device__ __forceinline__ v16bf ldB(const __bf16* __restrict__ Bf, int ks, int ntile, int lane) {
  return *(const v16bf*)(Bf + (size_t)(((ks * 16 + ntile) * 32 + lane) * 16));
}

// ---------------- WMMA GEMM: C[n,256] = A[n,256] @ W (pre-swizzled bf16) ----------------
// block = 256 threads = 8 waves; block tile 32 rows x 256 cols; wave tile 16x64.
// Software-pipelined K loop with double-buffered A/B fragments. act: 0=none, 1=tanh
__global__ __launch_bounds__(256)
void k_gemm256_wmma(const float* __restrict__ A, const __bf16* __restrict__ Bf,
                    const float* __restrict__ bias, float* __restrict__ C,
                    int nrows, int act) {
  int lane = threadIdx.x & 31;
  int wave = threadIdx.x >> 5;
  int rowbase = blockIdx.x * 32 + (wave >> 2) * 16;
  int ct0 = (wave & 3) * 4;            // first of 4 column tiles (16 cols each)
  int m = rowbase + (lane & 15);
  int khalf = lane >> 4;
  bool rowok = (m < nrows);
  const float* arow = A + (size_t)(rowok ? m : 0) * HDIM;

  v8f acc[4];
  #pragma unroll
  for (int t = 0; t < 4; ++t) acc[t] = (v8f)(0.0f);

  // prologue: fragments for ks = 0
  v16bf aC = ldA(arow, 0, khalf, rowok);
  v16bf bC[4];
  #pragma unroll
  for (int t = 0; t < 4; ++t) bC[t] = ldB(Bf, 0, ct0 + t, lane);

  // pipelined steps ks = 0..6: prefetch ks+1, then issue WMMAs for ks
  #pragma unroll
  for (int ks = 0; ks < 7; ++ks) {
    v16bf aN = ldA(arow, ks + 1, khalf, rowok);
    v16bf bN[4];
    #pragma unroll
    for (int t = 0; t < 4; ++t) bN[t] = ldB(Bf, ks + 1, ct0 + t, lane);
    #pragma unroll
    for (int t = 0; t < 4; ++t)
      acc[t] = __builtin_amdgcn_wmma_f32_16x16x32_bf16(
          false, aC, false, bC[t], (short)0, acc[t], false, false);
    aC = aN;
    #pragma unroll
    for (int t = 0; t < 4; ++t) bC[t] = bN[t];
  }
  // epilogue WMMAs: ks = 7
  #pragma unroll
  for (int t = 0; t < 4; ++t)
    acc[t] = __builtin_amdgcn_wmma_f32_16x16x32_bf16(
        false, aC, false, bC[t], (short)0, acc[t], false, false);

  #pragma unroll
  for (int t = 0; t < 4; ++t) {
    int col = (ct0 + t) * 16 + (lane & 15);
    float bv = bias ? bias[col] : 0.0f;
    #pragma unroll
    for (int r = 0; r < 8; ++r) {
      int row = rowbase + khalf * 8 + r;   // C/D layout: M = r + 8*(lane>>4)
      if (row < nrows) {
        float v = acc[t][r] + bv;
        if (act == 1) v = tanhf(v);
        C[(size_t)row * HDIM + col] = v;
      }
    }
  }
}

// ---------------- GAT edge passes ----------------
__global__ void k_edge_max(const float* __restrict__ a_s, const float* __restrict__ a_d,
                           const int* __restrict__ src, const int* __restrict__ dst,
                           float* __restrict__ mmax, int E) {
  int t = blockIdx.x * blockDim.x + threadIdx.x;
  if (t >= E * NHEAD) return;
  int e = t >> 2, h = t & 3;
  float x = a_s[src[e] * NHEAD + h] + a_d[dst[e] * NHEAD + h];
  x = x > 0.0f ? x : 0.2f * x;
  atomicMaxF(&mmax[dst[e] * NHEAD + h], x);
}

__global__ void k_edge_exp(const float* __restrict__ a_s, const float* __restrict__ a_d,
                           const int* __restrict__ src, const int* __restrict__ dst,
                           const float* __restrict__ mmax, float* __restrict__ den,
                           float* __restrict__ exb, int E) {
  int t = blockIdx.x * blockDim.x + threadIdx.x;
  if (t >= E * NHEAD) return;
  int e = t >> 2, h = t & 3;
  int d = dst[e];
  float x = a_s[src[e] * NHEAD + h] + a_d[d * NHEAD + h];
  x = x > 0.0f ? x : 0.2f * x;
  float ex = __expf(x - mmax[d * NHEAD + h]);
  exb[t] = ex;
  atomicAdd(&den[d * NHEAD + h], ex);
}

__global__ __launch_bounds__(256)
void k_edge_scatter(const float* __restrict__ hs, const int* __restrict__ src,
                    const int* __restrict__ dst, const float* __restrict__ exb,
                    const float* __restrict__ den, float* __restrict__ mout) {
  int e = blockIdx.x, c = threadIdx.x, h = c >> 6;
  int s = src[e], d = dst[e];
  float w = exb[e * NHEAD + h] / (den[d * NHEAD + h] + 1e-16f);
  atomicAdd(&mout[(size_t)d * HDIM + c], w * hs[(size_t)s * HDIM + c]);
}

// ---------------- attention pooling ----------------
__global__ void k_pool_max(const float* __restrict__ scores, const int* __restrict__ batch,
                           float* __restrict__ pm, int n) {
  int t = blockIdx.x * blockDim.x + threadIdx.x;
  if (t >= n * NHEAD) return;
  int row = t >> 2, h = t & 3;
  atomicMaxF(&pm[batch[row] * NHEAD + h], scores[t]);
}

__global__ void k_pool_expsum(float* __restrict__ scores, const int* __restrict__ batch,
                              const float* __restrict__ pm, float* __restrict__ pden, int n) {
  int t = blockIdx.x * blockDim.x + threadIdx.x;
  if (t >= n * NHEAD) return;
  int row = t >> 2, h = t & 3;
  int b = batch[row];
  float ex = __expf(scores[t] - pm[b * NHEAD + h]);
  scores[t] = ex;
  atomicAdd(&pden[b * NHEAD + h], ex);
}

__global__ void k_pool_wsum(const float* __restrict__ scores, const int* __restrict__ batch,
                            const float* __restrict__ pden, float* __restrict__ wsn, int n) {
  int row = blockIdx.x * blockDim.x + threadIdx.x;
  if (row >= n) return;
  int b = batch[row];
  float acc = 0.0f;
  for (int h = 0; h < NHEAD; ++h)
    acc += scores[row * NHEAD + h] / fmaxf(pden[b * NHEAD + h], 1e-10f);
  wsn[row] = acc;
}

__global__ __launch_bounds__(256)
void k_pool_scatter(const float* __restrict__ vals, const float* __restrict__ wsn,
                    const int* __restrict__ batch, float* __restrict__ pooled, int sect) {
  int n = blockIdx.x, c = threadIdx.x;
  atomicAdd(&pooled[(size_t)batch[n] * (3 * HDIM) + sect * HDIM + c],
            wsn[n] * vals[(size_t)n * HDIM + c]);
}

// ---------------- final heads: mu / logvar ----------------
__global__ void k_head(const float* __restrict__ pooled,
                       const float* __restrict__ muW, const float* __restrict__ mub,
                       const float* __restrict__ lvW, const float* __restrict__ lvb,
                       float* __restrict__ out) {
  int t = blockIdx.x * blockDim.x + threadIdx.x;
  if (t >= 2 * NGRAPH * LATD) return;
  int half = t / (NGRAPH * LATD);
  int rem = t % (NGRAPH * LATD);
  int g = rem / LATD, j = rem % LATD;
  const float* W = half ? lvW : muW;
  const float* b = half ? lvb : mub;
  const float* pr = pooled + (size_t)g * (3 * HDIM);
  float acc = b[j];
  for (int k = 0; k < 3 * HDIM; ++k) acc += pr[k] * 0.25f * W[k * LATD + j]; // /POOL_HEADS
  out[t] = acc;
}

// ---------------- host orchestration ----------------
static inline int cdiv(int a, int b) { return (a + b - 1) / b; }
static inline void fillf(float* p, float v, long n, hipStream_t s) {
  int blocks = (int)((n + 255) / 256); if (blocks > 4096) blocks = 4096;
  k_fill<<<blocks, 256, 0, s>>>(p, v, n);
}

extern "C" void kernel_launch(void* const* d_in, const int* in_sizes, int n_in,
                              void* d_out, int out_size, void* d_ws, size_t ws_size,
                              hipStream_t stream) {
  const float* x_vertex = (const float*)d_in[0];
  const float* x_edge   = (const float*)d_in[1];
  const float* x_face   = (const float*)d_in[2];
  const int* edge_type  = (const int*)d_in[3];
  const int* face_type  = (const int*)d_in[4];
  const int *ve_src=(const int*)d_in[5],  *ve_dst=(const int*)d_in[6];
  const int *ev_src=(const int*)d_in[7],  *ev_dst=(const int*)d_in[8];
  const int *ef_src=(const int*)d_in[9],  *ef_dst=(const int*)d_in[10];
  const int *fe_src=(const int*)d_in[11], *fe_dst=(const int*)d_in[12];
  const int *batch_v=(const int*)d_in[13], *batch_e=(const int*)d_in[14], *batch_f=(const int*)d_in[15];
  const float* ete=(const float*)d_in[16];
  const float* fte=(const float*)d_in[17];
  const float *vW=(const float*)d_in[18], *vb=(const float*)d_in[19], *vg=(const float*)d_in[20], *vbe=(const float*)d_in[21];
  const float *eW=(const float*)d_in[22], *eb=(const float*)d_in[23], *eg=(const float*)d_in[24], *ebe=(const float*)d_in[25];
  const float *fW=(const float*)d_in[26], *fb=(const float*)d_in[27], *fg=(const float*)d_in[28], *fbe=(const float*)d_in[29];
  const float* gWsrc=(const float*)d_in[30];
  const float* gWdst=(const float*)d_in[31];
  const float* gasrc=(const float*)d_in[32];
  const float* gadst=(const float*)d_in[33];
  const float* gbias=(const float*)d_in[34];
  const float* lng=(const float*)d_in[35];
  const float* lnb=(const float*)d_in[36];
  const float *pW1=(const float*)d_in[37], *pb1=(const float*)d_in[38];
  const float *pW2=(const float*)d_in[39], *pb2=(const float*)d_in[40];
  const float *pWv=(const float*)d_in[41], *pbv=(const float*)d_in[42];
  const float *muW=(const float*)d_in[43], *mub=(const float*)d_in[44];
  const float *lvW=(const float*)d_in[45], *lvb=(const float*)d_in[46];
  (void)in_sizes; (void)n_in; (void)out_size; (void)ws_size;

  float* ws = (float*)d_ws;
  size_t cur = 0;
  auto alloc = [&](size_t n) { float* p = ws + cur; cur += (n + 63) & ~(size_t)63; return p; };
  float* hv   = alloc((size_t)NVERT * HDIM);
  float* he   = alloc((size_t)NEDGE * HDIM);
  float* hf   = alloc((size_t)NFACE * HDIM);
  float* hs   = alloc((size_t)NEDGE * HDIM);   // max src rows; also pool "t" buffer
  float* mv   = alloc((size_t)NVERT * HDIM);
  float* me   = alloc((size_t)NEDGE * HDIM);   // also pool "values" buffer
  float* mf   = alloc((size_t)NFACE * HDIM);
  float* as_  = alloc((size_t)NEDGE * NHEAD);  // also pool scores
  float* ad_  = alloc((size_t)NEDGE * NHEAD);  // also pool per-node weight sum
  float* mmax = alloc((size_t)NEDGE * NHEAD);
  float* dden = alloc((size_t)NEDGE * NHEAD);
  float* exb  = alloc((size_t)EREL * NHEAD);
  float* pm   = alloc((size_t)NGRAPH * NHEAD);
  float* pden = alloc((size_t)NGRAPH * NHEAD);
  float* pooled = alloc((size_t)NGRAPH * 3 * HDIM);
  __bf16* wfrag = (__bf16*)alloc((size_t)18 * 65536 / 2); // 18 matrices * 65536 bf16
  float* cm   = alloc((size_t)24 * HDIM * NHEAD);         // collapsed attn mats

  // --- pre-swizzle all 256x256 weight matrices used by WMMA GEMMs ---
  for (int i = 0; i < 12; ++i)  // gat_Wsrc[l][r]
    k_swizzle_w<<<256, 256, 0, stream>>>(gWsrc + (size_t)i * 65536, wfrag + (size_t)i * 65536);
  for (int i = 0; i < 3; ++i)   // pool_W1[i]
    k_swizzle_w<<<256, 256, 0, stream>>>(pW1 + (size_t)i * 65536, wfrag + (size_t)(12 + i) * 65536);
  for (int i = 0; i < 3; ++i)   // pool_Wv[i]
    k_swizzle_w<<<256, 256, 0, stream>>>(pWv + (size_t)i * 65536, wfrag + (size_t)(15 + i) * 65536);
  // collapsed a_s / a_d matrices: cm[(l*4+r)*2 + {0=src,1=dst}]
  for (int i = 0; i < 12; ++i) {
    k_collapse_aw<<<4, 256, 0, stream>>>(gWsrc + (size_t)i * 65536, gasrc + (size_t)i * 256,
                                         cm + (size_t)(i * 2 + 0) * 1024);
    k_collapse_aw<<<4, 256, 0, stream>>>(gWdst + (size_t)i * 65536, gadst + (size_t)i * 256,
                                         cm + (size_t)(i * 2 + 1) * 1024);
  }

  // --- input projections ---
  k_proj_ln_relu<<<NVERT, 256, 0, stream>>>(x_vertex, 3, nullptr, nullptr, vW, vb, vg, vbe, hv);
  k_proj_ln_relu<<<NEDGE, 256, 0, stream>>>(x_edge, 6, ete, edge_type, eW, eb, eg, ebe, he);
  k_proj_ln_relu<<<NFACE, 256, 0, stream>>>(x_face, 13, fte, face_type, fW, fb, fg, fbe, hf);

  // relation table: 0: v->e, 1: e->v, 2: e->f, 3: f->e
  const float* srcH[4] = { hv, he, he, hf };
  const float* dstH[4] = { he, hv, hf, he };
  int ns[4] = { NVERT, NEDGE, NEDGE, NFACE };
  int nd[4] = { NEDGE, NVERT, NFACE, NEDGE };
  const int* esrc[4] = { ve_src, ev_src, ef_src, fe_src };
  const int* edst[4] = { ve_dst, ev_dst, ef_dst, fe_dst };
  float* mbuf[4] = { me, mv, mf, me };

  for (int l = 0; l < NLAYER; ++l) {
    fillf(mv, 0.0f, (long)NVERT * HDIM, stream);
    fillf(me, 0.0f, (long)NEDGE * HDIM, stream);
    fillf(mf, 0.0f, (long)NFACE * HDIM, stream);
    for (int r = 0; r < 4; ++r) {
      int idx = l * 4 + r;
      k_gemm256_wmma<<<cdiv(ns[r], 32), 256, 0, stream>>>(
          srcH[r], wfrag + (size_t)idx * 65536, nullptr, hs, ns[r], 0);
      k_thin_gemm4<<<cdiv(ns[r] * NHEAD, 256), 256, 0, stream>>>(
          srcH[r], cm + (size_t)(idx * 2 + 0) * 1024, nullptr, as_, ns[r]);
      k_thin_gemm4<<<cdiv(nd[r] * NHEAD, 256), 256, 0, stream>>>(
          dstH[r], cm + (size_t)(idx * 2 + 1) * 1024, nullptr, ad_, nd[r]);
      fillf(mmax, -INFINITY, (long)nd[r] * NHEAD, stream);
      fillf(dden, 0.0f, (long)nd[r] * NHEAD, stream);
      k_edge_max<<<cdiv(EREL * NHEAD, 256), 256, 0, stream>>>(as_, ad_, esrc[r], edst[r], mmax, EREL);
      k_edge_exp<<<cdiv(EREL * NHEAD, 256), 256, 0, stream>>>(as_, ad_, esrc[r], edst[r], mmax, dden, exb, EREL);
      k_edge_scatter<<<EREL, 256, 0, stream>>>(hs, esrc[r], edst[r], exb, dden, mbuf[r]);
    }
    k_update_ln_relu<<<NVERT, 256, 0, stream>>>(hv, mv, gbias + (size_t)(l * 4 + 1) * HDIM, nullptr,
                                                lng + (size_t)(l * 3 + 0) * HDIM, lnb + (size_t)(l * 3 + 0) * HDIM);
    k_update_ln_relu<<<NEDGE, 256, 0, stream>>>(he, me, gbias + (size_t)(l * 4 + 0) * HDIM,
                                                gbias + (size_t)(l * 4 + 3) * HDIM,
                                                lng + (size_t)(l * 3 + 1) * HDIM, lnb + (size_t)(l * 3 + 1) * HDIM);
    k_update_ln_relu<<<NFACE, 256, 0, stream>>>(hf, mf, gbias + (size_t)(l * 4 + 2) * HDIM, nullptr,
                                                lng + (size_t)(l * 3 + 2) * HDIM, lnb + (size_t)(l * 3 + 2) * HDIM);
  }

  // --- attention pooling ---
  fillf(pooled, 0.0f, (long)NGRAPH * 3 * HDIM, stream);
  const float* ph[3] = { hv, he, hf };
  int pn[3] = { NVERT, NEDGE, NFACE };
  const int* pbatch[3] = { batch_v, batch_e, batch_f };
  for (int i = 0; i < 3; ++i) {
    // t = tanh(h @ W1 + b1) -> hs ; vals = h @ Wv + bv -> me
    k_gemm256_wmma<<<cdiv(pn[i], 32), 256, 0, stream>>>(
        ph[i], wfrag + (size_t)(12 + i) * 65536, pb1 + (size_t)i * HDIM, hs, pn[i], 1);
    k_gemm256_wmma<<<cdiv(pn[i], 32), 256, 0, stream>>>(
        ph[i], wfrag + (size_t)(15 + i) * 65536, pbv + (size_t)i * HDIM, me, pn[i], 0);
    k_thin_gemm4<<<cdiv(pn[i] * NHEAD, 256), 256, 0, stream>>>(
        hs, pW2 + (size_t)i * HDIM * NHEAD, pb2 + (size_t)i * NHEAD, as_, pn[i]);
    fillf(pm, -INFINITY, (long)NGRAPH * NHEAD, stream);
    fillf(pden, 0.0f, (long)NGRAPH * NHEAD, stream);
    k_pool_max<<<cdiv(pn[i] * NHEAD, 256), 256, 0, stream>>>(as_, pbatch[i], pm, pn[i]);
    k_pool_expsum<<<cdiv(pn[i] * NHEAD, 256), 256, 0, stream>>>(as_, pbatch[i], pm, pden, pn[i]);
    k_pool_wsum<<<cdiv(pn[i], 256), 256, 0, stream>>>(as_, pbatch[i], pden, ad_, pn[i]);
    k_pool_scatter<<<pn[i], 256, 0, stream>>>(me, ad_, pbatch[i], pooled, i);
  }

  // --- heads ---
  k_head<<<cdiv(2 * NGRAPH * LATD, 256), 256, 0, stream>>>(pooled, muW, mub, lvW, lvb, (float*)d_out);
}
